// QattenMixer_91319594648353
// MI455X (gfx1250) — matrix operand order repaired and checked by reading, not compiled, not run on hardware
//
#include <hip/hip_runtime.h>

// ---------------- problem constants ----------------
#define NROWS 32768      // BS*T
#define SDIM  1600       // state dim
#define NB    1408       // fused hidden: 4*256 (q) + 128 (wh) + 256 (c)
#define NHEAD 4
#define Q1D   256
#define Q2D   64
#define AAG   32
#define UDIM  48
#define HD1D  128
#define CDIM  256

typedef __attribute__((ext_vector_type(16))) __bf16 v16bf;
typedef __attribute__((ext_vector_type(8)))  float  v8f;

union FragBF {
    v16bf v;
    uint4 u[2];
};

struct alignas(8) BF4 { __bf16 v[4]; };

// ---------------- CDNA5 async global->LDS staging ----------------
#if __has_builtin(__builtin_amdgcn_global_load_async_to_lds_b128) && \
    __has_builtin(__builtin_amdgcn_s_wait_asynccnt)
#define USE_ASYNC_LDS 1
#else
#define USE_ASYNC_LDS 0
#endif

// ---------------- CDNA5 Tensor Data Mover ----------------
#if __has_builtin(__builtin_amdgcn_tensor_load_to_lds) && \
    __has_builtin(__builtin_amdgcn_s_wait_tensorcnt)
#define HAVE_TDM 1
#else
#define HAVE_TDM 0
#endif

typedef int v4i_gv __attribute__((vector_size(16)));  // async builtin param pointee
typedef unsigned int tdm_v4u __attribute__((vector_size(16)));
typedef int tdm_v4i __attribute__((vector_size(16)));
typedef int tdm_v8i __attribute__((vector_size(32)));

__device__ __forceinline__ void copy_g2l_b128(const __bf16* g, __bf16* l) {
#if USE_ASYNC_LDS
    // global_load_async_to_lds_b128: LDS write engine, tracked by ASYNCcnt.
    __builtin_amdgcn_global_load_async_to_lds_b128(
        (__attribute__((address_space(1))) v4i_gv*)(uintptr_t)g,
        (__attribute__((address_space(3))) v4i_gv*)(uint32_t)(uintptr_t)l,
        0, 0);
#else
    *(uint4*)l = *(const uint4*)g;
#endif
}

__device__ __forceinline__ void lds_copy_fence() {
#if USE_ASYNC_LDS
    __builtin_amdgcn_s_wait_asynccnt(0);
#endif
    __syncthreads();
}

#if HAVE_TDM
// Issue one 2D TDM tile load: tile_h rows of tile_w bf16 elements, global row
// stride row_stride (elements), optionally padding the LDS destination every
// (2<<pad_interval_code) dwords by (pad_amount_code+1) dwords.
// D# layout per CDNA5 ISA 8.3 (group0) / 8.4 (group1); groups 2-3 zero (2D).
__device__ __forceinline__ void tdm_load_tile_bf16(const __bf16* gsrc, uint32_t lds_byte,
                                                   uint32_t tile_w, uint32_t tile_h,
                                                   uint32_t row_stride,
                                                   uint32_t pad_interval_code,
                                                   uint32_t pad_amount_code,
                                                   uint32_t pad_en) {
    uint64_t ga = (uint64_t)(uintptr_t)gsrc;
    tdm_v4u g0 = {
        1u,                                               // count=1, user mode, no gather
        lds_byte,                                         // lds_addr
        (uint32_t)ga,                                     // global_addr[31:0]
        (uint32_t)((ga >> 32) & 0x01FFFFFFu) | (2u << 30) // global_addr[56:32] | type=2
    };
    tdm_v8i g1 = {
        (int)((1u << 16) | (pad_en << 20) |
              (pad_interval_code << 22) | (pad_amount_code << 25)), // data_size=2B + pad cfg
        (int)((tile_w & 0xFFFFu) << 16),                  // tensor_dim0 lo16 (abar addr = 0)
        (int)(((tile_w >> 16) & 0xFFFFu) | ((tile_h & 0xFFFFu) << 16)), // dim0 hi | dim1 lo
        (int)(((tile_h >> 16) & 0xFFFFu) | ((tile_w & 0xFFFFu) << 16)), // dim1 hi | tile_dim0
        (int)(tile_h & 0xFFFFu),                          // tile_dim1 (tile_dim2 = 0)
        (int)row_stride,                                  // tensor_dim0_stride lo32
        0,                                                // stride hi16 | dim1_stride lo16
        0
    };
    tdm_v4i gz = {0, 0, 0, 0};
#if defined(__clang_major__) && (__clang_major__ >= 23)
    tdm_v8i gz8 = {0, 0, 0, 0, 0, 0, 0, 0};
    __builtin_amdgcn_tensor_load_to_lds(g0, g1, gz, gz, gz8, 0);
#else
    __builtin_amdgcn_tensor_load_to_lds(g0, g1, gz, gz, 0);
#endif
}
#endif

__device__ inline float wave_max32(float v) {
    #pragma unroll
    for (int m = 16; m > 0; m >>= 1) v = fmaxf(v, __shfl_xor(v, m, 32));
    return v;
}
__device__ inline float wave_sum32(float v) {
    #pragma unroll
    for (int m = 16; m > 0; m >>= 1) v += __shfl_xor(v, m, 32);
    return v;
}

// ---------------- pack kernels ----------------
__global__ __launch_bounds__(256) void pack_states_kernel(const float* __restrict__ src,
                                                          __bf16* __restrict__ dst, int n4) {
    int i = blockIdx.x * 256 + threadIdx.x;
    if (i >= n4) return;
    float4 f = ((const float4*)src)[i];
    BF4 o;
    o.v[0] = (__bf16)f.x; o.v[1] = (__bf16)f.y;
    o.v[2] = (__bf16)f.z; o.v[3] = (__bf16)f.w;
    *(BF4*)(dst + (size_t)i * 4) = o;
}

// WbigT[j][s] : j in [0,1408) output column, s in [0,1600) contraction dim.
__global__ __launch_bounds__(256) void pack_wbig_kernel(const float* __restrict__ Wq1,
                                                        const float* __restrict__ bq1,
                                                        const float* __restrict__ Wh1,
                                                        const float* __restrict__ bh1,
                                                        const float* __restrict__ Wc1,
                                                        const float* __restrict__ bc1,
                                                        __bf16* __restrict__ WbigT,
                                                        float* __restrict__ bbig) {
    int i = blockIdx.x * 256 + threadIdx.x;
    if (i >= NB * SDIM) return;
    int j = i / SDIM;
    int s = i - j * SDIM;
    float v;
    if (j < NHEAD * Q1D) {
        int h = j >> 8, q = j & 255;
        v = Wq1[((size_t)h * SDIM + s) * Q1D + q];
    } else if (j < NHEAD * Q1D + HD1D) {
        v = Wh1[(size_t)s * HD1D + (j - NHEAD * Q1D)];
    } else {
        v = Wc1[(size_t)s * CDIM + (j - NHEAD * Q1D - HD1D)];
    }
    WbigT[i] = (__bf16)v;
    if (s == 0) {
        float b;
        if (j < NHEAD * Q1D)              b = bq1[j];
        else if (j < NHEAD * Q1D + HD1D)  b = bh1[j - NHEAD * Q1D];
        else                              b = bc1[j - NHEAD * Q1D - HD1D];
        bbig[j] = b;
    }
}

// Wq2T[h][d][k] = Wq2[h][k][d]  (B transposed for contiguous fragments)
__global__ __launch_bounds__(256) void pack_wq2t_kernel(const float* __restrict__ Wq2,
                                                        __bf16* __restrict__ Wq2T) {
    int i = blockIdx.x * 256 + threadIdx.x;
    if (i >= NHEAD * Q2D * Q1D) return;
    int h = i >> 14;              // / (64*256)
    int d = (i >> 8) & 63;
    int k = i & 255;
    Wq2T[i] = (__bf16)Wq2[((size_t)h * Q1D + k) * Q2D + d];
}

// WkTE[h][u][d]: rows u<48 = Wk[h][u][:], row 48 = bk[h], rows 49..63 = 0.
// Transposed-B for p = Wk @ q2 with the bias dot appended as output col 48.
__global__ __launch_bounds__(256) void pack_wkte_kernel(const float* __restrict__ Wk,
                                                        const float* __restrict__ bk,
                                                        __bf16* __restrict__ WkTE) {
    int i = blockIdx.x * 256 + threadIdx.x;
    if (i >= NHEAD * 64 * 64) return;
    int h = i >> 12;
    int u = (i >> 6) & 63;
    int d = i & 63;
    float v;
    if (u < UDIM)       v = Wk[((size_t)h * UDIM + u) * Q2D + d];
    else if (u == UDIM) v = bk[h * Q2D + d];
    else                v = 0.f;
    WkTE[i] = (__bf16)v;
}

// ---------------- GEMM1: hid = relu(st @ Wbig + bbig), bf16 out ----------------
// block tile 128(M) x 128(N), 8 waves, wave tile 64x32, K-step 32,
// double-buffered LDS staged by the Tensor Data Mover (wave 0 drives DMA).
__global__ __launch_bounds__(256) void gemm1_kernel(const __bf16* __restrict__ stB,
                                                    const __bf16* __restrict__ WbigT,
                                                    const float* __restrict__ bbig,
                                                    __bf16* __restrict__ hidB) {
    __shared__ __bf16 a_s[2][128 * 40];   // [m][k], padded stride 40 (TDM pad 16dw->+4dw)
    __shared__ __bf16 b_s[2][128 * 40];   // [n][k] (B pre-transposed), padded

    const int tid   = threadIdx.x;
    const int nBase = blockIdx.x * 128;
    const int mBase = blockIdx.y * 128;
    const int wv    = tid >> 5;
    const int lane  = tid & 31;
    const int l15   = lane & 15;
    const int hf    = lane >> 4;
    const int wm    = (wv & 1) * 64;
    const int wn    = (wv >> 1) * 32;

#if HAVE_TDM
    auto fill = [&](int buf, int k0) {
        if (tid == 0) {
            // 128 rows x 32 bf16, row stride SDIM elems; pad every 16 dwords by 4 dwords
            tdm_load_tile_bf16(stB + (size_t)mBase * SDIM + k0,
                               (uint32_t)(uintptr_t)&a_s[buf][0],
                               32, 128, SDIM, 3, 3, 1);
            tdm_load_tile_bf16(WbigT + (size_t)nBase * SDIM + k0,
                               (uint32_t)(uintptr_t)&b_s[buf][0],
                               32, 128, SDIM, 3, 3, 1);
        }
    };
    auto fence = [&]() {
        if (tid == 0) __builtin_amdgcn_s_wait_tensorcnt(0);
        __syncthreads();
    };
#else
    auto fill = [&](int buf, int k0) {
        #pragma unroll
        for (int it = 0; it < 2; ++it) {
            int e   = it * 256 + tid;
            int row = e >> 2;
            int seg = (e & 3) * 8;
            copy_g2l_b128(stB   + (size_t)(mBase + row) * SDIM + k0 + seg,
                          &a_s[buf][row * 40 + seg]);
            copy_g2l_b128(WbigT + (size_t)(nBase + row) * SDIM + k0 + seg,
                          &b_s[buf][row * 40 + seg]);
        }
    };
    auto fence = [&]() { lds_copy_fence(); };
#endif

    v8f zero8;
    #pragma unroll
    for (int i = 0; i < 8; ++i) zero8[i] = 0.f;
    v8f acc[4][2];
    #pragma unroll
    for (int i = 0; i < 4; ++i)
        #pragma unroll
        for (int j = 0; j < 2; ++j) acc[i][j] = zero8;

    fill(0, 0);
    fence();

    int cur = 0;
    for (int k0 = 0; k0 < SDIM; k0 += 32) {
        if (k0 + 32 < SDIM) fill(cur ^ 1, k0 + 32);   // DMA next tile into LDS

        const __bf16* as = a_s[cur];
        const __bf16* bs = b_s[cur];
        FragBF fa[4], fb[2];
        #pragma unroll
        for (int mt = 0; mt < 4; ++mt) {
            int m = wm + mt * 16 + l15;
            fa[mt].u[0] = *(const uint4*)(as + m * 40 + hf * 8);
            fa[mt].u[1] = *(const uint4*)(as + m * 40 + 16 + hf * 8);
        }
        #pragma unroll
        for (int nt = 0; nt < 2; ++nt) {
            int c = wn + nt * 16 + l15;
            fb[nt].u[0] = *(const uint4*)(bs + c * 40 + hf * 8);
            fb[nt].u[1] = *(const uint4*)(bs + c * 40 + 16 + hf * 8);
        }
        #pragma unroll
        for (int mt = 0; mt < 4; ++mt)
            #pragma unroll
            for (int nt = 0; nt < 2; ++nt)
                acc[mt][nt] = __builtin_amdgcn_wmma_f32_16x16x32_bf16(
                    false, fa[mt].v, false, fb[nt].v, (short)0, acc[mt][nt], false, false);

        fence();
        cur ^= 1;
    }

    #pragma unroll
    for (int mt = 0; mt < 4; ++mt)
        #pragma unroll
        for (int nt = 0; nt < 2; ++nt)
            #pragma unroll
            for (int r = 0; r < 8; ++r) {
                int rowG = mBase + wm + mt * 16 + r + hf * 8;
                int colG = nBase + wn + nt * 16 + l15;
                float v = acc[mt][nt][r] + bbig[colG];
                v = v > 0.f ? v : 0.f;
                hidB[(size_t)rowG * NB + colG] = (__bf16)v;
            }
}

// ---------------- GEMM2 + p: q2 = hid_q @ Wq2 + bq2; pExt = q2 @ [Wk^T | bk | 0] ----------------
// grid (NROWS/64, H), 128 threads = 4 waves, each wave owns 16 rows.
// Staging uses the async global->LDS path (ASYNCcnt).
__global__ __launch_bounds__(128) void gemm2p_kernel(const __bf16* __restrict__ hidB,
                                                     const __bf16* __restrict__ Wq2T,
                                                     const float* __restrict__ bq2,
                                                     const __bf16* __restrict__ WkTE,
                                                     float* __restrict__ pExt) {
    __shared__ __bf16 aT[2][64 * 40];
    __shared__ __bf16 bT[2][64 * 40];
    __shared__ __bf16 q2s[64 * 72];
    __shared__ __bf16 wkT[64 * 72];

    const int tid     = threadIdx.x;
    const int h       = blockIdx.y;
    const int rowBase = blockIdx.x * 64;
    const int wv      = tid >> 5;
    const int lane    = tid & 31;
    const int l15     = lane & 15;
    const int hf      = lane >> 4;

    auto fillAB = [&](int buf, int k0) {
        #pragma unroll
        for (int it = 0; it < 2; ++it) {
            int e   = it * 128 + tid;
            int row = e >> 2;
            int seg = (e & 3) * 8;
            copy_g2l_b128(hidB + (size_t)(rowBase + row) * NB + h * Q1D + k0 + seg,
                          &aT[buf][row * 40 + seg]);
            copy_g2l_b128(Wq2T + ((size_t)(h * 64 + row)) * Q1D + k0 + seg,
                          &bT[buf][row * 40 + seg]);
        }
    };

    // stage WkTE (64x64) once (completion covered by first fence)
    #pragma unroll
    for (int it = 0; it < 4; ++it) {
        int e   = it * 128 + tid;
        int row = e >> 3;
        int seg = (e & 7) * 8;
        copy_g2l_b128(WkTE + (size_t)h * 4096 + row * 64 + seg, &wkT[row * 72 + seg]);
    }

    v8f zero8;
    #pragma unroll
    for (int i = 0; i < 8; ++i) zero8[i] = 0.f;
    v8f acc[4];
    #pragma unroll
    for (int i = 0; i < 4; ++i) acc[i] = zero8;

    const int m = wv * 16 + l15;

    fillAB(0, 0);
    lds_copy_fence();

    int cur = 0;
    for (int ks = 0; ks < 8; ++ks) {
        int k0 = ks * 32;
        if (ks < 7) fillAB(cur ^ 1, k0 + 32);

        const __bf16* as = aT[cur];
        const __bf16* bs = bT[cur];
        FragBF fa;
        fa.u[0] = *(const uint4*)(as + m * 40 + hf * 8);
        fa.u[1] = *(const uint4*)(as + m * 40 + 16 + hf * 8);
        #pragma unroll
        for (int nt = 0; nt < 4; ++nt) {
            int c = nt * 16 + l15;
            FragBF fb;
            fb.u[0] = *(const uint4*)(bs + c * 40 + hf * 8);
            fb.u[1] = *(const uint4*)(bs + c * 40 + 16 + hf * 8);
            acc[nt] = __builtin_amdgcn_wmma_f32_16x16x32_bf16(
                false, fa.v, false, fb.v, (short)0, acc[nt], false, false);
        }

        lds_copy_fence();
        cur ^= 1;
    }

    // epilogue: +bq2 and stage q2 (bf16) as A operand for phase 2
    #pragma unroll
    for (int nt = 0; nt < 4; ++nt)
        #pragma unroll
        for (int r = 0; r < 8; ++r) {
            int row_l = wv * 16 + r + hf * 8;
            int col   = nt * 16 + l15;
            float v = acc[nt][r] + bq2[h * Q2D + col];
            q2s[row_l * 72 + col] = (__bf16)v;
        }
    __syncthreads();

    // phase 2: p_ext(16x64) = q2(16x64) @ wkT^T ; col 48 carries q2.bk
    v8f acc2[4];
    #pragma unroll
    for (int i = 0; i < 4; ++i) acc2[i] = zero8;

    #pragma unroll
    for (int ks2 = 0; ks2 < 2; ++ks2) {
        int k0 = ks2 * 32;
        FragBF fa2;
        fa2.u[0] = *(const uint4*)(q2s + m * 72 + k0 + hf * 8);
        fa2.u[1] = *(const uint4*)(q2s + m * 72 + k0 + 16 + hf * 8);
        #pragma unroll
        for (int nt = 0; nt < 4; ++nt) {
            int c = nt * 16 + l15;
            FragBF fb2;
            fb2.u[0] = *(const uint4*)(wkT + c * 72 + k0 + hf * 8);
            fb2.u[1] = *(const uint4*)(wkT + c * 72 + k0 + 16 + hf * 8);
            acc2[nt] = __builtin_amdgcn_wmma_f32_16x16x32_bf16(
                false, fa2.v, false, fb2.v, (short)0, acc2[nt], false, false);
        }
    }

    #pragma unroll
    for (int nt = 0; nt < 4; ++nt)
        #pragma unroll
        for (int r = 0; r < 8; ++r) {
            int rowG = rowBase + wv * 16 + r + hf * 8;
            int col  = nt * 16 + l15;
            pExt[((size_t)h * NROWS + rowG) * 64 + col] = acc2[nt][r];
        }
}

// ---------------- finalize: one wave (32 lanes == 32 agents) per row ----------------
__global__ __launch_bounds__(256) void finalize_kernel(const float* __restrict__ states,
                                                       const float* __restrict__ agent_qs,
                                                       const __bf16* __restrict__ hidB,
                                                       const float* __restrict__ pExt,
                                                       const float* __restrict__ Wh2,
                                                       const float* __restrict__ bh2,
                                                       const float* __restrict__ Wc2,
                                                       const float* __restrict__ bc2,
                                                       float* __restrict__ out) {
    const int wv   = threadIdx.x >> 5;
    const int lane = threadIdx.x & 31;
    const int n    = blockIdx.x * 8 + wv;

    // lane = agent index; us[a] = states[n, a*48 : a*48+48]
    const float* usp = states + (size_t)n * SDIM + lane * UDIM;
    float u[UDIM];
    #pragma unroll
    for (int i = 0; i < UDIM; ++i) u[i] = usp[i];
    const float aqv = agent_qs[n * AAG + lane];

    float qh[NHEAD];
    #pragma unroll
    for (int h = 0; h < NHEAD; ++h) {
        const float* pv = pExt + ((size_t)h * NROWS + n) * 64;
        float d = pv[UDIM];                // q2 . bk
        #pragma unroll
        for (int i = 0; i < UDIM; ++i) d += u[i] * pv[i];
        d *= 0.125f;                       // 1/sqrt(Q2)
        float mx = wave_max32(d);
        float e  = __expf(d - mx);
        float s  = wave_sum32(e);
        qh[h] = wave_sum32(aqv * (e / s));
    }

    const __bf16* hid = hidB + (size_t)n * NB;
    float wh[NHEAD];
    #pragma unroll
    for (int h = 0; h < NHEAD; ++h) {
        float p = 0.f;
        #pragma unroll
        for (int j = 0; j < 4; ++j) {
            int i = lane + j * 32;
            p += (float)hid[NHEAD * Q1D + i] * Wh2[i * NHEAD + h];
        }
        wh[h] = fabsf(wave_sum32(p) + bh2[h]);
    }

    float cp = 0.f;
    #pragma unroll
    for (int j = 0; j < 8; ++j) {
        int i = lane + j * 32;
        cp += (float)hid[NHEAD * Q1D + HD1D + i] * Wc2[i];
    }
    float c = wave_sum32(cp) + bc2[0];

    if (lane == 0)
        out[n] = qh[0] * wh[0] + qh[1] * wh[1] + qh[2] * wh[2] + qh[3] * wh[3] + c;
}

// ---------------- host ----------------
extern "C" void kernel_launch(void* const* d_in, const int* in_sizes, int n_in,
                              void* d_out, int out_size, void* d_ws, size_t ws_size,
                              hipStream_t stream) {
    const float* agent_qs = (const float*)d_in[0];
    const float* states   = (const float*)d_in[1];
    const float* Wq1 = (const float*)d_in[2];
    const float* bq1 = (const float*)d_in[3];
    const float* Wq2 = (const float*)d_in[4];
    const float* bq2 = (const float*)d_in[5];
    const float* Wk  = (const float*)d_in[6];
    const float* bk  = (const float*)d_in[7];
    const float* Wh1 = (const float*)d_in[8];
    const float* bh1 = (const float*)d_in[9];
    const float* Wh2 = (const float*)d_in[10];
    const float* bh2 = (const float*)d_in[11];
    const float* Wc1 = (const float*)d_in[12];
    const float* bc1 = (const float*)d_in[13];
    const float* Wc2 = (const float*)d_in[14];
    const float* bc2 = (const float*)d_in[15];
    float* out = (float*)d_out;

    // workspace layout
    char* w = (char*)d_ws;
    size_t cur = 0;
    auto take = [&](size_t bytes) {
        void* p = w + cur;
        cur = (cur + bytes + 255) & ~(size_t)255;
        return p;
    };
    __bf16* stB   = (__bf16*)take((size_t)NROWS * SDIM * 2);
    __bf16* WbigT = (__bf16*)take((size_t)NB * SDIM * 2);
    float*  bbig  = (float*) take((size_t)NB * 4);
    __bf16* Wq2T  = (__bf16*)take((size_t)NHEAD * Q2D * Q1D * 2);
    __bf16* WkTE  = (__bf16*)take((size_t)NHEAD * 64 * 64 * 2);
    __bf16* hidB  = (__bf16*)take((size_t)NROWS * NB * 2);
    float*  pExt  = (float*) take((size_t)NHEAD * NROWS * 64 * 4);
    (void)ws_size; (void)in_sizes; (void)n_in; (void)out_size;

    // pack / convert
    {
        int n4 = NROWS * SDIM / 4;
        pack_states_kernel<<<(n4 + 255) / 256, 256, 0, stream>>>(states, stB, n4);
        int nw = NB * SDIM;
        pack_wbig_kernel<<<(nw + 255) / 256, 256, 0, stream>>>(Wq1, bq1, Wh1, bh1, Wc1, bc1,
                                                               WbigT, bbig);
        pack_wq2t_kernel<<<(NHEAD * Q2D * Q1D + 255) / 256, 256, 0, stream>>>(Wq2, Wq2T);
        pack_wkte_kernel<<<(NHEAD * 64 * 64 + 255) / 256, 256, 0, stream>>>(Wk, bk, WkTE);
    }

    // fused big GEMM with bias+ReLU epilogue (TDM-staged tiles)
    gemm1_kernel<<<dim3(NB / 128, NROWS / 128), 256, 0, stream>>>(stB, WbigT, bbig, hidB);

    // q2 + p (with bk column) per head (async-LDS staged tiles)
    gemm2p_kernel<<<dim3(NROWS / 64, NHEAD), 128, 0, stream>>>(hidB, Wq2T, bq2, WkTE, pExt);

    // attention softmax + head mixing + constant head
    finalize_kernel<<<NROWS / 8, 256, 0, stream>>>(states, agent_qs, hidB, pExt,
                                                   Wh2, bh2, Wc2, bc2, out);
}
